// MakeMoEPreStage_41558103556279
// MI455X (gfx1250) — matrix operand order repaired and checked by reading, not compiled, not run on hardware
//
#include <hip/hip_runtime.h>
#include <hip/hip_bf16.h>
#include <stdint.h>

// ---------------------------------------------------------------------------
// MakeMoE pre-stage block for MI455X (gfx1250, wave32, WMMA + async-to-LDS).
// B=4 T=2048 D=1024 H=8 HD=128 E=8 TOPK=2.
// ---------------------------------------------------------------------------

typedef __attribute__((ext_vector_type(16))) _Float16 v16h;
typedef __attribute__((ext_vector_type(8)))  _Float16 v8h;
typedef __attribute__((ext_vector_type(4)))  _Float16 v4h;
typedef __attribute__((ext_vector_type(8)))  float    v8f;
typedef __attribute__((ext_vector_type(4)))  float    v4f;

#define BATCH 4
#define TSEQ  2048
#define DMODEL 1024
#define NHEAD 8
#define HD    128
#define NEXP  8
#define NROWS (BATCH * TSEQ)          // 8192

#define GM 128                         // block M tile
#define GNB 128                        // block N tile
#define GK 32                          // K step (= WMMA K)
#define PITCH 40                       // LDS row pitch in halves (80B, 20 banks)

// ---- WMMA fragment helpers (ISA 7.12.2 layouts, wave32) --------------------

// A fragment 16x32 f16: m = mbase + (lane&15); k chunks of 8 at
// kk + (lane>>4)*8 and kk + 16 + (lane>>4)*8.
__device__ __forceinline__ v16h load_a_frag(const _Float16* A, int lda,
                                            int mbase, int kk, int lane) {
  const int m = mbase + (lane & 15);
  const int g = lane >> 4;
  const _Float16* p = A + (size_t)m * lda + kk + g * 8;
  v8h lo = *reinterpret_cast<const v8h*>(p);
  v8h hi = *reinterpret_cast<const v8h*>(p + 16);
  v16h a;
#pragma unroll
  for (int i = 0; i < 8; ++i) { a[i] = lo[i]; a[i + 8] = hi[i]; }
  return a;
}

// B fragment 32x16 f16 from N-major Bt[n][k]: n = nbase + (lane&15),
// k chunks of 8 at kk + (lane>>4)*8 and kk + 16 + (lane>>4)*8 (2x16B).
__device__ __forceinline__ v16h load_b_frag(const _Float16* Bt, int ldb,
                                            int nbase, int kk, int lane) {
  const int n = nbase + (lane & 15);
  const int g = lane >> 4;
  const _Float16* p = Bt + (size_t)n * ldb + kk + g * 8;
  v8h lo = *reinterpret_cast<const v8h*>(p);
  v8h hi = *reinterpret_cast<const v8h*>(p + 16);
  v16h b;
#pragma unroll
  for (int i = 0; i < 8; ++i) { b[i] = lo[i]; b[i + 8] = hi[i]; }
  return b;
}

__device__ __forceinline__ v8f wmma_f32(v16h a, v16h b, v8f c) {
  return __builtin_amdgcn_wmma_f32_16x16x32_f16(false, a, false, b,
                                                (short)0, c, false, false);
}

// ---- gfx1250 async global->LDS copy (ASYNCcnt-tracked, no VGPR data) ------
__device__ __forceinline__ void async_ld16(const _Float16* lds_dst,
                                           const _Float16* gsrc) {
  const unsigned loff = (unsigned)(uintptr_t)lds_dst;   // flat[31:0] == LDS off
  const uint64_t gaddr = (uint64_t)(uintptr_t)gsrc;
  asm volatile("global_load_async_to_lds_b128 %0, %1, off"
               :: "v"(loff), "v"(gaddr) : "memory");
}
__device__ __forceinline__ void wait_async_pending() {   // next stage in flight
  asm volatile("s_wait_asynccnt 0x4" ::: "memory");
}
__device__ __forceinline__ void wait_async_all() {       // final stage
  asm volatile("s_wait_asynccnt 0x0" ::: "memory");
}

// Stage one GMxGK A slice + GNBxGK B slice into LDS (4 async b128 / thread).
__device__ __forceinline__ void stage_tiles(const _Float16* __restrict__ A, int lda,
                                            const _Float16* __restrict__ Bt, int ldb,
                                            int mblk, int nblk, int kk,
                                            _Float16* Ab, _Float16* Bb, int t) {
  int c = t, row = c >> 2, sub = c & 3;               // A chunks 0..255
  async_ld16(Ab + row * PITCH + sub * 8,
             A + (size_t)(mblk + row) * lda + kk + sub * 8);
  c = t + 256; row = c >> 2; sub = c & 3;             // A chunks 256..511
  async_ld16(Ab + row * PITCH + sub * 8,
             A + (size_t)(mblk + row) * lda + kk + sub * 8);
  c = t; row = c >> 2; sub = c & 3;                   // B chunks 0..255
  async_ld16(Bb + row * PITCH + sub * 8,
             Bt + (size_t)(nblk + row) * ldb + kk + sub * 8);
  c = t + 256; row = c >> 2; sub = c & 3;             // B chunks 256..511
  async_ld16(Bb + row * PITCH + sub * 8,
             Bt + (size_t)(nblk + row) * ldb + kk + sub * 8);
}

// ---------------------------------------------------------------------------
// K1: weight prep -> f16, N-major transposed for contiguous B fragments.
// ---------------------------------------------------------------------------
__global__ __launch_bounds__(256)
void prep_weights_k(const float* __restrict__ Wq, const float* __restrict__ Wk,
                    const float* __restrict__ Wv, const float* __restrict__ Wo,
                    _Float16* __restrict__ wqkvT, _Float16* __restrict__ woT) {
  const size_t i = (size_t)blockIdx.x * 256 + threadIdx.x;
  const size_t NQKV = (size_t)3072 * 1024;
  if (i < NQKV) {
    const int n = (int)(i >> 10);
    const int c = (int)(i & 1023);
    const int qkv = n >> 10, h = (n >> 7) & 7, d = n & 127;
    const float* W = (qkv == 0) ? Wq : (qkv == 1) ? Wk : Wv;
    wqkvT[i] = (_Float16)W[((size_t)h * 1024 + c) * 128 + d];
  } else {
    const size_t j = i - NQKV;
    const int n = (int)(j >> 10), c = (int)(j & 1023);
    woT[j] = (_Float16)Wo[(size_t)c * 1024 + n];
  }
}

// ---------------------------------------------------------------------------
// K2: embedding + LayerNorm1 (block per row).
// ---------------------------------------------------------------------------
__global__ __launch_bounds__(256)
void embed_ln1_k(const int* __restrict__ idx, const float* __restrict__ tok,
                 const float* __restrict__ pos, const float* __restrict__ g1,
                 const float* __restrict__ b1, float* __restrict__ xout,
                 _Float16* __restrict__ xh) {
  __shared__ float ssum[8], ssq[8];
  const int row = blockIdx.x;
  const int t = row & (TSEQ - 1);
  const int id = idx[row];
  const int d0 = threadIdx.x * 4;
  v4f tv = *reinterpret_cast<const v4f*>(tok + (size_t)id * DMODEL + d0);
  v4f pv = *reinterpret_cast<const v4f*>(pos + (size_t)t * DMODEL + d0);
  v4f xv;
  float s = 0.f, q = 0.f;
#pragma unroll
  for (int i = 0; i < 4; ++i) { xv[i] = tv[i] + pv[i]; s += xv[i]; q += xv[i] * xv[i]; }
#pragma unroll
  for (int m = 1; m < 32; m <<= 1) { s += __shfl_xor(s, m, 32); q += __shfl_xor(q, m, 32); }
  if ((threadIdx.x & 31) == 0) { ssum[threadIdx.x >> 5] = s; ssq[threadIdx.x >> 5] = q; }
  __syncthreads();
  s = 0.f; q = 0.f;
#pragma unroll
  for (int i = 0; i < 8; ++i) { s += ssum[i]; q += ssq[i]; }
  const float mu = s * (1.0f / DMODEL);
  const float var = q * (1.0f / DMODEL) - mu * mu;
  const float rstd = rsqrtf(var + 1e-5f);
  v4f gv = *reinterpret_cast<const v4f*>(g1 + d0);
  v4f bv = *reinterpret_cast<const v4f*>(b1 + d0);
  v4h hv;
#pragma unroll
  for (int i = 0; i < 4; ++i) hv[i] = (_Float16)((xv[i] - mu) * rstd * gv[i] + bv[i]);
  *reinterpret_cast<v4f*>(xout + (size_t)row * DMODEL + d0) = xv;
  *reinterpret_cast<v4h*>(xh + (size_t)row * DMODEL + d0) = hv;
}

// ---------------------------------------------------------------------------
// K3: fused QKV GEMM, block-cooperative, async double-buffered LDS staging.
// Block tile 128x128, 8 waves (4Mx2N), each wave 32x64 -> 8 WMMA per K-slice.
// ---------------------------------------------------------------------------
__global__ __launch_bounds__(256)
void gemm_qkv_k(const _Float16* __restrict__ xh, const _Float16* __restrict__ wqkvT,
                _Float16* __restrict__ qb, _Float16* __restrict__ kb,
                _Float16* __restrict__ vtb) {
  __shared__ _Float16 Ab[2][GM * PITCH];    // 2 x 10240B
  __shared__ _Float16 Bb[2][GNB * PITCH];   // 2 x 10240B
  const int t = threadIdx.x, lane = t & 31, wid = t >> 5;
  const int mblk = (blockIdx.x / 24) * GM;      // 3072/128 = 24 N-blocks
  const int nblk = (blockIdx.x % 24) * GNB;
  const int wm = (wid >> 1) * 32, wn = (wid & 1) * 64;
  v8f acc[2][4] = {};

  stage_tiles(xh, DMODEL, wqkvT, DMODEL, mblk, nblk, 0, Ab[0], Bb[0], t);
  for (int kk = 0; kk < DMODEL; kk += GK) {
    const int cur = (kk >> 5) & 1;
    if (kk + GK < DMODEL) {
      stage_tiles(xh, DMODEL, wqkvT, DMODEL, mblk, nblk, kk + GK,
                  Ab[cur ^ 1], Bb[cur ^ 1], t);
      wait_async_pending();   // retire current stage, next stays in flight
    } else {
      wait_async_all();
    }
    __syncthreads();          // publish all waves' slices
    v16h a0 = load_a_frag(Ab[cur], PITCH, wm,      0, lane);
    v16h a1 = load_a_frag(Ab[cur], PITCH, wm + 16, 0, lane);
#pragma unroll
    for (int j = 0; j < 4; ++j) {
      v16h bj = load_b_frag(Bb[cur], PITCH, wn + j * 16, 0, lane);
      acc[0][j] = wmma_f32(a0, bj, acc[0][j]);
      acc[1][j] = wmma_f32(a1, bj, acc[1][j]);
    }
    __syncthreads();          // readers done before buffer is re-staged
  }

#pragma unroll
  for (int i = 0; i < 2; ++i)
#pragma unroll
    for (int j = 0; j < 4; ++j) {
      const int n = nblk + wn + j * 16 + (lane & 15);
      const int qkv = n >> 10, h = (n >> 7) & 7, d = n & 127;
#pragma unroll
      for (int r = 0; r < 8; ++r) {
        const int m = mblk + wm + i * 16 + (lane >> 4) * 8 + r;
        const int bb = m >> 11, tt = m & (TSEQ - 1);
        const _Float16 val = (_Float16)acc[i][j][r];
        if (qkv == 0)
          qb[(((size_t)(bb * NHEAD + h)) * TSEQ + tt) * HD + d] = val;
        else if (qkv == 1)
          kb[(((size_t)(bb * NHEAD + h)) * TSEQ + tt) * HD + d] = val;
        else
          vtb[(((size_t)(bb * NHEAD + h)) * HD + d) * TSEQ + tt] = val;
      }
    }
}

// ---------------------------------------------------------------------------
// K4: causal flash attention. One wave per (b,h, 16-query tile).
// ---------------------------------------------------------------------------
__global__ __launch_bounds__(256)
void attention_k(const _Float16* __restrict__ qb, const _Float16* __restrict__ kb,
                 const _Float16* __restrict__ vtb, _Float16* __restrict__ attnb) {
  __shared__ _Float16 pstage[8][16 * 32];
  const int lane = threadIdx.x & 31;
  const int wid = threadIdx.x >> 5;
  const int wg = blockIdx.x * 8 + wid;         // 4096 waves total
  const int bh = wg >> 7;
  const int qt = wg & 127;
  const int qbase = qt * 16;
  const int b = bh >> 3, h = bh & 7;
  const _Float16* Q  = qb  + (size_t)bh * TSEQ * HD;
  const _Float16* Km = kb  + (size_t)bh * TSEQ * HD;
  const _Float16* Vt = vtb + (size_t)bh * HD * TSEQ;
  const int g = lane >> 4;
  const int ln = lane & 15;

  v16h qa[4];
#pragma unroll
  for (int c = 0; c < 4; ++c) qa[c] = load_a_frag(Q, HD, qbase, c * 32, lane);

  float mrun[8], lrun[8];
  v8f O[8] = {};
#pragma unroll
  for (int r = 0; r < 8; ++r) { mrun[r] = -3.0e38f; lrun[r] = 0.f; }

  const float scale = 0.03125f;   // D**-0.5 (reference scales by n_embed)

  for (int sbase = 0; sbase <= qbase + 15; sbase += 32) {
    if (sbase + 32 <= qbase + 15)   // prefetch next K tile (global_prefetch_b8)
      __builtin_prefetch(Km + (size_t)(sbase + 32 + lane) * HD, 0, 0);
    v8f S0 = {}, S1 = {};
#pragma unroll
    for (int c = 0; c < 4; ++c) {
      v16h kb0 = load_b_frag(Km, HD, sbase,      c * 32, lane);
      v16h kb1 = load_b_frag(Km, HD, sbase + 16, c * 32, lane);
      S0 = wmma_f32(qa[c], kb0, S0);
      S1 = wmma_f32(qa[c], kb1, S1);
    }
    const int key0 = sbase + ln, key1 = sbase + 16 + ln;
    float p0[8], p1[8], alpha[8];
#pragma unroll
    for (int r = 0; r < 8; ++r) {
      const int qrow = qbase + g * 8 + r;
      float s0 = (key0 <= qrow) ? S0[r] * scale : -3.0e38f;
      float s1 = (key1 <= qrow) ? S1[r] * scale : -3.0e38f;
      float rmax = fmaxf(s0, s1);
#pragma unroll
      for (int mm = 1; mm < 16; mm <<= 1) rmax = fmaxf(rmax, __shfl_xor(rmax, mm, 32));
      const float mnew = fmaxf(mrun[r], rmax);
      alpha[r] = __expf(mrun[r] - mnew);
      p0[r] = __expf(s0 - mnew);
      p1[r] = __expf(s1 - mnew);
      float rsum = p0[r] + p1[r];
#pragma unroll
      for (int mm = 1; mm < 16; mm <<= 1) rsum += __shfl_xor(rsum, mm, 32);
      lrun[r] = lrun[r] * alpha[r] + rsum;
      mrun[r] = mnew;
    }
#pragma unroll
    for (int j = 0; j < 8; ++j)
#pragma unroll
      for (int r = 0; r < 8; ++r) O[j][r] *= alpha[r];

    // stage P (C-layout) -> LDS -> reload as A-fragment (wave-private tile)
    _Float16* ps = pstage[wid];
#pragma unroll
    for (int r = 0; r < 8; ++r) {
      ps[(g * 8 + r) * 32 + ln]      = (_Float16)p0[r];
      ps[(g * 8 + r) * 32 + 16 + ln] = (_Float16)p1[r];
    }
    asm volatile("s_wait_dscnt 0x0" ::: "memory");   // wave-local LDS RAW fence
    v16h pa = load_a_frag(ps, 32, 0, 0, lane);
#pragma unroll
    for (int j = 0; j < 8; ++j) {
      v16h vb = load_b_frag(Vt, TSEQ, j * 16, sbase, lane);
      O[j] = wmma_f32(pa, vb, O[j]);
    }
  }

  float inv[8];
#pragma unroll
  for (int r = 0; r < 8; ++r) inv[r] = 1.0f / lrun[r];
#pragma unroll
  for (int j = 0; j < 8; ++j)
#pragma unroll
    for (int r = 0; r < 8; ++r) {
      const int tt = qbase + g * 8 + r;
      attnb[((size_t)(b * TSEQ + tt)) * DMODEL + h * HD + j * 16 + ln] =
          (_Float16)(O[j][r] * inv[r]);
    }
}

// ---------------------------------------------------------------------------
// K5: out projection + bias + residual (same block-cooperative scheme).
// x += attn @ Wo + bo   (f32, in d_out)
// ---------------------------------------------------------------------------
__global__ __launch_bounds__(256)
void gemm_out_k(const _Float16* __restrict__ attnb, const _Float16* __restrict__ woT,
                const float* __restrict__ bo, float* __restrict__ x) {
  __shared__ _Float16 Ab[2][GM * PITCH];
  __shared__ _Float16 Bb[2][GNB * PITCH];
  const int t = threadIdx.x, lane = t & 31, wid = t >> 5;
  const int mblk = (blockIdx.x / 8) * GM;       // 1024/128 = 8 N-blocks
  const int nblk = (blockIdx.x % 8) * GNB;
  const int wm = (wid >> 1) * 32, wn = (wid & 1) * 64;
  v8f acc[2][4] = {};

  stage_tiles(attnb, DMODEL, woT, DMODEL, mblk, nblk, 0, Ab[0], Bb[0], t);
  for (int kk = 0; kk < DMODEL; kk += GK) {
    const int cur = (kk >> 5) & 1;
    if (kk + GK < DMODEL) {
      stage_tiles(attnb, DMODEL, woT, DMODEL, mblk, nblk, kk + GK,
                  Ab[cur ^ 1], Bb[cur ^ 1], t);
      wait_async_pending();
    } else {
      wait_async_all();
    }
    __syncthreads();
    v16h a0 = load_a_frag(Ab[cur], PITCH, wm,      0, lane);
    v16h a1 = load_a_frag(Ab[cur], PITCH, wm + 16, 0, lane);
#pragma unroll
    for (int j = 0; j < 4; ++j) {
      v16h bj = load_b_frag(Bb[cur], PITCH, wn + j * 16, 0, lane);
      acc[0][j] = wmma_f32(a0, bj, acc[0][j]);
      acc[1][j] = wmma_f32(a1, bj, acc[1][j]);
    }
    __syncthreads();
  }

#pragma unroll
  for (int i = 0; i < 2; ++i)
#pragma unroll
    for (int j = 0; j < 4; ++j) {
      const int n = nblk + wn + j * 16 + (lane & 15);
      const float bias = bo[n];
#pragma unroll
      for (int r = 0; r < 8; ++r) {
        const int m = mblk + wm + i * 16 + (lane >> 4) * 8 + r;
        x[(size_t)m * DMODEL + n] += acc[i][j][r] + bias;
      }
    }
}

// ---------------------------------------------------------------------------
// K6: LayerNorm2 -> x_norm (f32 output) + f16 copy for the router.
// ---------------------------------------------------------------------------
__global__ __launch_bounds__(256)
void ln2_k(const float* __restrict__ xin, const float* __restrict__ g2,
           const float* __restrict__ b2, float* __restrict__ yout,
           _Float16* __restrict__ yh) {
  __shared__ float ssum[8], ssq[8];
  const int row = blockIdx.x;
  const int d0 = threadIdx.x * 4;
  v4f xv = *reinterpret_cast<const v4f*>(xin + (size_t)row * DMODEL + d0);
  float s = 0.f, q = 0.f;
#pragma unroll
  for (int i = 0; i < 4; ++i) { s += xv[i]; q += xv[i] * xv[i]; }
#pragma unroll
  for (int m = 1; m < 32; m <<= 1) { s += __shfl_xor(s, m, 32); q += __shfl_xor(q, m, 32); }
  if ((threadIdx.x & 31) == 0) { ssum[threadIdx.x >> 5] = s; ssq[threadIdx.x >> 5] = q; }
  __syncthreads();
  s = 0.f; q = 0.f;
#pragma unroll
  for (int i = 0; i < 8; ++i) { s += ssum[i]; q += ssq[i]; }
  const float mu = s * (1.0f / DMODEL);
  const float var = q * (1.0f / DMODEL) - mu * mu;
  const float rstd = rsqrtf(var + 1e-5f);
  v4f gv = *reinterpret_cast<const v4f*>(g2 + d0);
  v4f bv = *reinterpret_cast<const v4f*>(b2 + d0);
  v4f yv; v4h hv;
#pragma unroll
  for (int i = 0; i < 4; ++i) {
    yv[i] = (xv[i] - mu) * rstd * gv[i] + bv[i];
    hv[i] = (_Float16)yv[i];
  }
  *reinterpret_cast<v4f*>(yout + (size_t)row * DMODEL + d0) = yv;
  *reinterpret_cast<v4h*>(yh + (size_t)row * DMODEL + d0) = hv;
}

// ---------------------------------------------------------------------------
// K7: noisy top-2 router. One wave per (b,t) row. E=8 experts.
// ---------------------------------------------------------------------------
__device__ __forceinline__ unsigned pcg_hash(unsigned v) {
  v = v * 747796405u + 2891336453u;
  v = ((v >> ((v >> 28u) + 4u)) ^ v) * 277803737u;
  return (v >> 22u) ^ v;
}

__global__ __launch_bounds__(256)
void router_k(const _Float16* __restrict__ xh2, const float* __restrict__ Wr,
              const float* __restrict__ br, const float* __restrict__ Wn,
              const float* __restrict__ bn, float* __restrict__ rp,
              int* __restrict__ ridx) {
  const int lane = threadIdx.x & 31;
  const int row = blockIdx.x * 8 + (threadIdx.x >> 5);
  float aR[8] = {}, aN[8] = {};
  for (int c = lane; c < DMODEL; c += 32) {
    const float xv = (float)xh2[(size_t)row * DMODEL + c];
    v8f wr = *reinterpret_cast<const v8f*>(Wr + (size_t)c * 8);
    v8f wn = *reinterpret_cast<const v8f*>(Wn + (size_t)c * 8);
#pragma unroll
    for (int e = 0; e < 8; ++e) { aR[e] += xv * wr[e]; aN[e] += xv * wn[e]; }
  }
#pragma unroll
  for (int m = 1; m < 32; m <<= 1)
#pragma unroll
    for (int e = 0; e < 8; ++e) {
      aR[e] += __shfl_xor(aR[e], m, 32);
      aN[e] += __shfl_xor(aN[e], m, 32);
    }
  if (lane == 0) {
    float noisy[8];
#pragma unroll
    for (int e = 0; e < 8; ++e) {
      const float lg = aR[e] + br[e];
      const float nl = aN[e] + bn[e];
      const float sp = (nl > 20.f) ? nl : log1pf(__expf(nl));
      const unsigned h1 = pcg_hash((unsigned)(row * 8 + e) + 0x2A42A42Au);
      const unsigned h2 = pcg_hash(h1 ^ 0x9E3779B9u);
      const float u1 = (float)(h1 >> 8) * (1.0f / 16777216.0f) + 1e-7f;
      const float u2 = (float)(h2 >> 8) * (1.0f / 16777216.0f);
      const float gauss = sqrtf(-2.0f * __logf(u1)) * __cosf(6.2831853f * u2);
      noisy[e] = lg + gauss * sp;
    }
    int i0 = 0;
#pragma unroll
    for (int e = 1; e < 8; ++e) if (noisy[e] > noisy[i0]) i0 = e;
    int i1 = (i0 == 0) ? 1 : 0;
#pragma unroll
    for (int e = 0; e < 8; ++e) if (e != i0 && noisy[e] > noisy[i1]) i1 = e;
    const float ep = __expf(noisy[i1] - noisy[i0]);
    const float pTop = 1.0f / (1.0f + ep);
    const float pSec = ep * pTop;
#pragma unroll
    for (int e = 0; e < 8; ++e)
      rp[(size_t)row * 8 + e] = (e == i0) ? pTop : ((e == i1) ? pSec : 0.0f);
    ridx[(size_t)row * 2 + 0] = i0;
    ridx[(size_t)row * 2 + 1] = i1;
  }
}

// ---------------------------------------------------------------------------
// kernel_launch
// inputs: 0 idx, 1 tok_table, 2 pos_table, 3 Wq, 4 Wk, 5 Wv, 6 Wo, 7 bo,
//         8 g1, 9 b1, 10 g2, 11 b2, 12 Wr, 13 br, 14 Wn, 15 bn
// d_out: [x_norm f32][router_probs f32][indices i32][x f32]
// ---------------------------------------------------------------------------
extern "C" void kernel_launch(void* const* d_in, const int* in_sizes, int n_in,
                              void* d_out, int out_size, void* d_ws, size_t ws_size,
                              hipStream_t stream) {
  const int*   idx = (const int*)d_in[0];
  const float* tok = (const float*)d_in[1];
  const float* pos = (const float*)d_in[2];
  const float* Wq  = (const float*)d_in[3];
  const float* Wk  = (const float*)d_in[4];
  const float* Wv  = (const float*)d_in[5];
  const float* Wo  = (const float*)d_in[6];
  const float* bo  = (const float*)d_in[7];
  const float* g1  = (const float*)d_in[8];
  const float* b1  = (const float*)d_in[9];
  const float* g2  = (const float*)d_in[10];
  const float* b2  = (const float*)d_in[11];
  const float* Wr  = (const float*)d_in[12];
  const float* br  = (const float*)d_in[13];
  const float* Wn  = (const float*)d_in[14];
  const float* bn  = (const float*)d_in[15];

  float* out_xnorm = (float*)d_out;                                   // 8192*1024
  float* out_rp    = out_xnorm + (size_t)NROWS * DMODEL;              // 8192*8
  int*   out_idx   = (int*)(out_rp + (size_t)NROWS * NEXP);           // 8192*2
  float* out_x     = (float*)(out_idx + (size_t)NROWS * 2);           // 8192*1024

  _Float16* xh    = (_Float16*)d_ws;
  _Float16* wqkvT = xh + (size_t)NROWS * DMODEL;          // 3072*1024
  _Float16* woT   = wqkvT + (size_t)3072 * 1024;          // 1024*1024
  _Float16* qb    = woT + (size_t)1024 * 1024;            // [B,H,T,HD]
  _Float16* kb    = qb + (size_t)NROWS * DMODEL;
  _Float16* vtb   = kb + (size_t)NROWS * DMODEL;          // [B,H,HD,T]
  _Float16* attnb = vtb + (size_t)NROWS * DMODEL;         // [B,T,D]
  _Float16* xn2h  = attnb + (size_t)NROWS * DMODEL;

  prep_weights_k<<<(3072 * 1024 + 1024 * 1024) / 256, 256, 0, stream>>>(
      Wq, Wk, Wv, Wo, wqkvT, woT);
  embed_ln1_k<<<NROWS, 256, 0, stream>>>(idx, tok, pos, g1, b1, out_x, xh);
  gemm_qkv_k<<<(NROWS / GM) * (3072 / GNB), 256, 0, stream>>>(xh, wqkvT, qb, kb, vtb);
  attention_k<<<4096 / 8, 256, 0, stream>>>(qb, kb, vtb, attnb);
  gemm_out_k<<<(NROWS / GM) * (DMODEL / GNB), 256, 0, stream>>>(attnb, woT, bo, out_x);
  ln2_k<<<NROWS, 256, 0, stream>>>(out_x, g2, b2, out_xnorm, xn2h);
  router_k<<<NROWS / 8, 256, 0, stream>>>(xn2h, Wr, br, Wn, bn, out_rp, out_idx);
}
